// LinearGLUMoELayer_1924145348663
// MI455X (gfx1250) — compile-verified, matrix operand order checked by reading, not run on hardware
//
#include <hip/hip_runtime.h>
#include <hip/hip_bf16.h>
#include <math.h>

#define Tn 4096
#define In 4096
#define Hn 1376
#define On 4096
#define En 8
#define Hp 1408                 // H padded to multiple of 64
#define EHp (En * Hp)           // 11264 = 176 * 64
#define LDP 72                  // LDS row pitch (ushorts): 144B rows, 16B-aligned everywhere

typedef __attribute__((ext_vector_type(16))) __bf16       v16bf;
typedef __attribute__((ext_vector_type(8)))  float        v8f;
typedef __attribute__((ext_vector_type(8)))  unsigned int v8u;
typedef __attribute__((ext_vector_type(4)))  unsigned int su4;
typedef __attribute__((ext_vector_type(8)))  unsigned int su8;

#define S_WAIT_ASYNC(n) asm volatile("s_wait_asynccnt %0" :: "i"(n) : "memory")

// CDNA5 async global->LDS copy (16B per lane), tracked by ASYNCcnt.
__device__ __forceinline__ void async_b128(unsigned lds_off, const void* gptr) {
  asm volatile("global_load_async_to_lds_b128 %0, %1, off"
               :: "v"(lds_off), "v"((unsigned long long)(uintptr_t)gptr)
               : "memory");
}

// ---- Tensor Data Mover: 2D tile (64 cols x `rows` rows of bf16) global -> LDS.
// D# group1: data_size=2B; LDS padding 4 DWORDs after every 32 DWORDs (one 128B row)
// -> rows land at 144B pitch == LDP ushorts, matching make_frag.
__device__ __forceinline__ su8 tdm_desc_g1(unsigned rows, unsigned stride_elems) {
  su8 g1;
  g1[0] = (1u << 16)        // data_size = 1 -> 2 bytes
        | (1u << 20)        // pad_enable
        | (4u << 22)        // pad_interval: 2^(4+1) = 32 DWORDs (128B row)
        | (3u << 25);       // pad_amount: 3+1 = 4 DWORDs (16B)
  g1[1] = (64u & 0xFFFFu) << 16;                    // tensor_dim0[15:0] = 64
  g1[2] = 0u | ((rows & 0xFFFFu) << 16);            // tensor_dim0[31:16]=0 | tensor_dim1[15:0]
  g1[3] = 0u | (64u << 16);                         // tensor_dim1[31:16]=0 | tile_dim0 = 64
  g1[4] = rows & 0xFFFFu;                           // tile_dim1 = rows ; tile_dim2 = 0
  g1[5] = stride_elems;                             // tensor_dim0_stride[31:0]
  g1[6] = 0u;                                       // stride0[47:32] | stride1[15:0]
  g1[7] = 0u;                                       // stride1[47:16]
  return g1;
}

__device__ __forceinline__ void tdm_load_2d(unsigned lds_off, const void* gaddr, su8 g1) {
  unsigned long long ga = (unsigned long long)(uintptr_t)gaddr;
  su4 g0;
  g0[0] = 1u;                                       // count=1, user descriptor
  g0[1] = lds_off;                                  // lds_addr
  g0[2] = (unsigned)ga;                             // global_addr[31:0]
  g0[3] = (unsigned)(ga >> 32) | (2u << 30);        // global_addr[56:32] | type=2 (image)
  asm volatile("tensor_load_to_lds %0, %1" :: "s"(g0), "s"(g1) : "memory");
}

__device__ __forceinline__ unsigned short f2bf(float f) {
  unsigned int u = __builtin_bit_cast(unsigned int, f);
  u += 0x7FFFu + ((u >> 16) & 1u);   // round-to-nearest-even
  return (unsigned short)(u >> 16);
}

// 16x32 bf16 fragment from LDS tile (row-major, pitch LDP).
// ISA 7.12.2: VGPR j<4 -> K = 8*half + 2j ; j>=4 -> K = 16 + 8*half + 2(j-4)
__device__ __forceinline__ v16bf make_frag(const unsigned short* s, int row, int half) {
  const unsigned short* p = s + row * LDP;
  v8u u;
#pragma unroll
  for (int j = 0; j < 8; ++j) {
    int kj = (j < 4) ? (8 * half + 2 * j) : (16 + 8 * half + 2 * (j - 4));
    u[j] = *(const unsigned int*)(p + kj);
  }
  return __builtin_bit_cast(v16bf, u);
}

// ---------------- fp32 -> bf16 conversion ----------------
__global__ __launch_bounds__(256) void cvt_f32_bf16(const float* __restrict__ src,
                                                    unsigned short* __restrict__ dst, int n) {
  int i = blockIdx.x * blockDim.x + threadIdx.x;
  int stride = gridDim.x * blockDim.x;
  for (; i < n; i += stride) dst[i] = f2bf(src[i]);
}

// w_down fp32 [E*O][Hn]  ->  bf16 [E*O][Hp] with zero pad columns
__global__ __launch_bounds__(256) void cvt_pad_wd(const float* __restrict__ src,
                                                  unsigned short* __restrict__ dst) {
  const size_t r = blockIdx.x;           // 0 .. E*On-1
  for (int h = threadIdx.x; h < Hp; h += 256)
    dst[r * Hp + h] = (h < Hn) ? f2bf(src[r * Hn + h]) : (unsigned short)0;
}

// zero the pad columns of interw [T][EHp]
__global__ __launch_bounds__(256) void zero_pad_interw(unsigned short* __restrict__ iw) {
  int idx = blockIdx.x * 256 + threadIdx.x;      // 4096*8*32 = 1,048,576
  int h = idx & 31;
  int e = (idx >> 5) & 7;
  int t = idx >> 8;
  iw[(size_t)t * EHp + e * Hp + Hn + h] = 0;
}

// ---------------- gate: top-2 softmax -> dense routing matrix ----------------
__global__ __launch_bounds__(256) void gate_kernel(const float* __restrict__ x,
                                                   const float* __restrict__ gw,
                                                   float* __restrict__ dense) {
  const int wave = threadIdx.x >> 5, lane = threadIdx.x & 31;
  const int t = blockIdx.x * 8 + wave;
  const float* xp = x + (size_t)t * In;
  float acc[En];
#pragma unroll
  for (int e = 0; e < En; ++e) acc[e] = 0.f;
  for (int i = lane; i < In; i += 32) {
    float xv = xp[i];
#pragma unroll
    for (int e = 0; e < En; ++e) acc[e] += xv * gw[e * In + i];
  }
#pragma unroll
  for (int e = 0; e < En; ++e) {
#pragma unroll
    for (int off = 16; off > 0; off >>= 1) acc[e] += __shfl_xor(acc[e], off, 32);
  }
  if (lane == 0) {
    int i0 = 0; float m0 = acc[0];
#pragma unroll
    for (int e = 1; e < En; ++e) if (acc[e] > m0) { m0 = acc[e]; i0 = e; }
    int i1 = 0; float m1 = -3.0e38f;
#pragma unroll
    for (int e = 0; e < En; ++e) if (e != i0 && acc[e] > m1) { m1 = acc[e]; i1 = e; }
    float e1 = __expf(m1 - m0);
    float s0 = 1.f / (1.f + e1);
    float s1 = e1 * s0;
#pragma unroll
    for (int e = 0; e < En; ++e) dense[t * En + e] = 0.f;
    dense[t * En + i0] = s0;
    dense[t * En + i1] = s1;
  }
}

// ---------------- balance loss: cv^2(importance) + cv^2(load) ----------------
__global__ __launch_bounds__(256) void gate_loss_kernel(const float* __restrict__ dense,
                                                        float* __restrict__ out_loss) {
  __shared__ float simp[256][En];
  __shared__ float sld[256][En];
  __shared__ float rimp[En], rld[En];
  const int tid = threadIdx.x;
  float imp[En], ld[En];
#pragma unroll
  for (int e = 0; e < En; ++e) { imp[e] = 0.f; ld[e] = 0.f; }
  for (int t = tid; t < Tn; t += 256) {
#pragma unroll
    for (int e = 0; e < En; ++e) {
      float v = dense[t * En + e];
      imp[e] += v;
      ld[e] += (v > 0.f) ? 1.f : 0.f;
    }
  }
#pragma unroll
  for (int e = 0; e < En; ++e) { simp[tid][e] = imp[e]; sld[tid][e] = ld[e]; }
  __syncthreads();
  if (tid < En) {
    float si = 0.f, sl = 0.f;
    for (int i = 0; i < 256; ++i) { si += simp[i][tid]; sl += sld[i][tid]; }
    rimp[tid] = si; rld[tid] = sl;
  }
  __syncthreads();
  if (tid == 0) {
    float mi = 0.f, ml = 0.f;
#pragma unroll
    for (int e = 0; e < En; ++e) { mi += rimp[e]; ml += rld[e]; }
    mi /= (float)En; ml /= (float)En;
    float vi = 0.f, vl = 0.f;
#pragma unroll
    for (int e = 0; e < En; ++e) {
      float d1 = rimp[e] - mi; vi += d1 * d1;
      float d2 = rld[e] - ml;  vl += d2 * d2;
    }
    vi /= (float)(En - 1); vl /= (float)(En - 1);
    out_loss[0] = 0.01f * (vi / (mi * mi + 1e-10f) + vl / (ml * ml + 1e-10f));
  }
}

// ---------------- B1: interw[t, e*Hp+h] = dense[t,e]*silu(x@Wg^T+bg)*(x@Wu^T+bu)
// One expert per block-z; 64 tokens x 32 h-cols; BK=64, async double-buffered LDS.
__global__ __launch_bounds__(256) void moe_glu_kernel(
    const unsigned short* __restrict__ xb,     // [T][I] bf16
    const unsigned short* __restrict__ wg,     // [E][H][I] bf16
    const unsigned short* __restrict__ wu,     // [E][H][I] bf16
    const float* __restrict__ bgp,             // [E][H]
    const float* __restrict__ bup,             // [E][H]
    const float* __restrict__ dense,           // [T][E]
    unsigned short* __restrict__ interw) {     // [T][EHp] bf16
  __shared__ __align__(16) unsigned short As[2][64 * LDP];
  __shared__ __align__(16) unsigned short Gs[2][32 * LDP];
  __shared__ __align__(16) unsigned short Us[2][32 * LDP];
  const int e  = blockIdx.z;
  const int m0 = blockIdx.y * 64;
  const int h0 = blockIdx.x * 32;
  const int tid = threadIdx.x;
  const int lane = tid & 31, wave = tid >> 5;
  const int wm = wave & 3, wn = wave >> 2;
  const int half = lane >> 4, l15 = lane & 15;

  const int xr = tid >> 2, xc = (tid & 3) * 16;  // 64 rows x 4 chunks of 16 ushorts
  const int wr = tid >> 3, wc = (tid & 7) * 8;   // 32 rows x 8 chunks of 8 ushorts

  const unsigned short* xrow = xb + (size_t)(m0 + xr) * In + xc;
  const unsigned short* grow = wg + ((size_t)e * Hn + h0 + wr) * In + wc;
  const unsigned short* urow = wu + ((size_t)e * Hn + h0 + wr) * In + wc;

  unsigned axo[2], ago[2], auo[2];
#pragma unroll
  for (int b = 0; b < 2; ++b) {
    axo[b] = (unsigned)(uintptr_t)&As[b][xr * LDP + xc];
    ago[b] = (unsigned)(uintptr_t)&Gs[b][wr * LDP + wc];
    auo[b] = (unsigned)(uintptr_t)&Us[b][wr * LDP + wc];
  }

  auto issue = [&](int b, int k0) {
    async_b128(axo[b],      xrow + k0);
    async_b128(axo[b] + 16, xrow + k0 + 8);
    async_b128(ago[b],      grow + k0);
    async_b128(auo[b],      urow + k0);
  };

  v8f cg = {}; v8f cu = {};
  const int NC = In / 64;
  issue(0, 0);
  for (int c = 0; c < NC; ++c) {
    if (c + 1 < NC) { issue((c + 1) & 1, (c + 1) * 64); S_WAIT_ASYNC(4); }
    else            { S_WAIT_ASYNC(0); }
    __syncthreads();                       // all waves' chunk-c tiles in LDS
    const unsigned short* Ab = As[c & 1];
    const unsigned short* Gb = Gs[c & 1];
    const unsigned short* Ub = Us[c & 1];
#pragma unroll
    for (int s = 0; s < 2; ++s) {
      v16bf a  = make_frag(Ab + s * 32, wm * 16 + l15, half);
      v16bf fg = make_frag(Gb + s * 32, wn * 16 + l15, half);
      v16bf fu = make_frag(Ub + s * 32, wn * 16 + l15, half);
      cg = __builtin_amdgcn_wmma_f32_16x16x32_bf16(false, a, false, fg, (short)0, cg, false, false);
      cu = __builtin_amdgcn_wmma_f32_16x16x32_bf16(false, a, false, fu, (short)0, cu, false, false);
    }
    __syncthreads();                       // buffer may be overwritten next iter
  }

  const int hcol = h0 + wn * 16 + l15;
  const float bgv = bgp[e * Hn + hcol];
  const float buv = bup[e * Hn + hcol];
#pragma unroll
  for (int j = 0; j < 8; ++j) {
    int t = m0 + wm * 16 + j + 8 * half;   // C layout: M = j + 8*half, N = lane&15
    float g = cg[j] + bgv;
    float u = cu[j] + buv;
    float sg = g / (1.f + __expf(-g));     // silu
    float w = dense[t * En + e];
    interw[(size_t)t * EHp + e * Hp + hcol] = f2bf(sg * u * w);
  }
}

// ---------------- B2: y = interw[T, EHp] @ WdAll[O, EHp]^T + dense @ b_down ----
// Pure GEMM over K=11264 (pad cols zero); BK=64; tiles fetched by the Tensor Data
// Mover (wave 0 -> A tile, wave 1 -> B tile), double-buffered on TENSORcnt.
__global__ __launch_bounds__(256) void down_proj_kernel(
    const unsigned short* __restrict__ iw,     // [T][EHp] bf16
    const unsigned short* __restrict__ wdp,    // [E][O][Hp] bf16 (padded)
    const float* __restrict__ bdp,             // [E][O]
    const float* __restrict__ dense,           // [T][E]
    float* __restrict__ y) {                   // [T][O]
  __shared__ __align__(16) unsigned short As[2][64 * LDP];
  __shared__ __align__(16) unsigned short Bs[2][64 * LDP];
  const int m0 = blockIdx.y * 64;
  const int n0 = blockIdx.x * 64;
  const int tid = threadIdx.x;
  const int lane = tid & 31, wave = tid >> 5;
  const int wm = wave & 3, wn = wave >> 2;
  const int half = lane >> 4, l15 = lane & 15;

  const su8 g1A = tdm_desc_g1(64, EHp);   // 64x64 tile, row stride EHp elems
  const su8 g1B = tdm_desc_g1(64, Hp);    // 64x64 tile, row stride Hp elems
  unsigned aoff[2], boff[2];
#pragma unroll
  for (int b = 0; b < 2; ++b) {
    aoff[b] = (unsigned)(uintptr_t)&As[b][0];
    boff[b] = (unsigned)(uintptr_t)&Bs[b][0];
  }

  auto issue = [&](int b, int kk, int ee, int hh) {
    if (wave == 0)
      tdm_load_2d(aoff[b], iw + (size_t)m0 * EHp + kk, g1A);
    else if (wave == 1)
      tdm_load_2d(boff[b], wdp + ((size_t)ee * On + n0) * Hp + hh, g1B);
  };

  v8f c0 = {}; v8f c1 = {};
  const int NC = EHp / 64;                 // 176
  issue(0, 0, 0, 0);
  int en = 0, hn = 64;                     // (e, h) of next chunk to issue
  for (int c = 0; c < NC; ++c) {
    if (c + 1 < NC) {
      issue((c + 1) & 1, (c + 1) * 64, en, hn);
      if (wave < 2) __builtin_amdgcn_s_wait_tensorcnt((short)1);  // chunk c retired
      hn += 64;
      if (hn == Hp) { hn = 0; ++en; }
    } else {
      if (wave < 2) __builtin_amdgcn_s_wait_tensorcnt((short)0);
    }
    __syncthreads();                       // chunk-c tiles visible to all waves
    const unsigned short* Ab = As[c & 1];
    const unsigned short* Bb = Bs[c & 1];
#pragma unroll
    for (int s = 0; s < 2; ++s) {
      v16bf a  = make_frag(Ab + s * 32, wm * 16 + l15, half);
      v16bf b0 = make_frag(Bb + s * 32, wn * 32 + l15, half);
      v16bf b1 = make_frag(Bb + s * 32, wn * 32 + 16 + l15, half);
      c0 = __builtin_amdgcn_wmma_f32_16x16x32_bf16(false, a, false, b0, (short)0, c0, false, false);
      c1 = __builtin_amdgcn_wmma_f32_16x16x32_bf16(false, a, false, b1, (short)0, c1, false, false);
    }
    __syncthreads();                       // buffer may be overwritten next iter
  }

#pragma unroll
  for (int j = 0; j < 8; ++j) {
    int t = m0 + wm * 16 + j + 8 * half;
    const float* dr = dense + t * En;
    float cc0 = c0[j], cc1 = c1[j];
#pragma unroll
    for (int s = 0; s < 2; ++s) {
      int o = n0 + wn * 32 + s * 16 + l15;
      float bias = 0.f;
#pragma unroll
      for (int e2 = 0; e2 < En; ++e2) bias += dr[e2] * bdp[e2 * On + o];
      y[(size_t)t * On + o] = (s == 0 ? cc0 : cc1) + bias;
    }
  }
}

extern "C" void kernel_launch(void* const* d_in, const int* in_sizes, int n_in,
                              void* d_out, int out_size, void* d_ws, size_t ws_size,
                              hipStream_t stream) {
  (void)in_sizes; (void)n_in; (void)out_size; (void)ws_size;
  const float* x   = (const float*)d_in[0];
  const float* gw  = (const float*)d_in[1];
  const float* wgf = (const float*)d_in[2];
  const float* bg  = (const float*)d_in[3];
  const float* wuf = (const float*)d_in[4];
  const float* bu  = (const float*)d_in[5];
  const float* wdf = (const float*)d_in[6];
  const float* bd  = (const float*)d_in[7];
  float* y = (float*)d_out;   // T*O floats, then 1 loss float

  char* ws = (char*)d_ws;
  size_t off = 0;
  auto alloc = [&](size_t bytes) {
    char* p = ws + off;
    off += (bytes + 255) & ~(size_t)255;
    return p;
  };
  float* dense          = (float*)alloc(sizeof(float) * Tn * En);
  unsigned short* xb    = (unsigned short*)alloc(2ull * Tn * In);
  unsigned short* wgb   = (unsigned short*)alloc(2ull * En * Hn * In);
  unsigned short* wub   = (unsigned short*)alloc(2ull * En * Hn * In);
  unsigned short* wdb   = (unsigned short*)alloc(2ull * En * On * Hp);   // padded
  unsigned short* iwbuf = (unsigned short*)alloc(2ull * Tn * EHp);       // padded

  cvt_f32_bf16<<<4096, 256, 0, stream>>>(x,   xb,  Tn * In);
  cvt_f32_bf16<<<8192, 256, 0, stream>>>(wgf, wgb, En * Hn * In);
  cvt_f32_bf16<<<8192, 256, 0, stream>>>(wuf, wub, En * Hn * In);
  cvt_pad_wd<<<En * On, 256, 0, stream>>>(wdf, wdb);
  zero_pad_interw<<<4096, 256, 0, stream>>>(iwbuf);

  gate_kernel<<<Tn / 8, 256, 0, stream>>>(x, gw, dense);
  gate_loss_kernel<<<1, 256, 0, stream>>>(dense, y + (size_t)Tn * On);

  moe_glu_kernel<<<dim3(Hn / 32, Tn / 64, En), 256, 0, stream>>>(
      xb, wgb, wub, bg, bu, dense, iwbuf);
  down_proj_kernel<<<dim3(On / 64, Tn / 64), 256, 0, stream>>>(
      iwbuf, wdb, bd, dense, y);
}
